// myGCN_63788854280505
// MI455X (gfx1250) — matrix operand order repaired and checked by reading, not compiled, not run on hardware
//
#include <hip/hip_runtime.h>

typedef __attribute__((ext_vector_type(16))) _Float16 v16h;
typedef __attribute__((ext_vector_type(8)))  _Float16 v8h;
typedef __attribute__((ext_vector_type(8)))  float    v8f;

#define DH 128
#define ROWS 64   // rows of A per block (4 row-tiles of 16)

__device__ __forceinline__ float eluf(float x) {
  return x > 0.0f ? x : (__expf(x) - 1.0f);
}

// ---------------- elementwise zero ----------------
__global__ void k_zero(float* __restrict__ p, long n) {
  long i = (long)blockIdx.x * blockDim.x + threadIdx.x;
  if (i < n) p[i] = 0.0f;
}

// ---------------- degree accumulation (buffer pre-zeroed) ----------------
__global__ void k_deg(const int* __restrict__ dst, float* __restrict__ deg, int E) {
  int e = blockIdx.x * blockDim.x + threadIdx.x;
  if (e < E) unsafeAtomicAdd(&deg[dst[e]], 1.0f);
}

// ---------------- deg -> rsqrt(deg + 1) in place ----------------
__global__ void k_dinv(float* __restrict__ d, int N) {
  int i = blockIdx.x * blockDim.x + threadIdx.x;
  if (i < N) d[i] = rsqrtf(d[i] + 1.0f);
}

// ---------------- f32 W[k][c] -> f16 transposed Wt[c][k] ----------------
__global__ void k_w2t(const float* __restrict__ w, _Float16* __restrict__ wt, int n) {
  int i = blockIdx.x * blockDim.x + threadIdx.x;
  if (i < n) {
    int c = i >> 7, k = i & 127;
    wt[i] = (_Float16)w[k * DH + c];
  }
}

// ---------------- WMMA GEMM: out[N,128] = act(A[N,128] @ W[128,128] + bias) ----------------
// Wt is f16 column-major [c][k]. 256 threads = 8 waves; block computes ROWS x 128;
// wave w owns cols [16w,16w+16) and iterates 4 row-tiles. act: 0 = none, 1 = ELU.
__global__ void __launch_bounds__(256)
k_gemm(const float* __restrict__ A, const _Float16* __restrict__ Wt,
       const float* __restrict__ bias, float* __restrict__ out, int N, int act) {
  __shared__ _Float16 sA[ROWS][136];   // [row][k], padded (272B stride)
  __shared__ _Float16 sW[128][136];    // [col][k], padded

  const int tid  = threadIdx.x;
  const int row0 = blockIdx.x * ROWS;

  // stage Wt (128x128 f16 = 32KB), 16B chunks -> ds_store_b128
  const float4* wv = (const float4*)Wt;   // 8 halves per float4
  for (int i = tid; i < 2048; i += 256) {
    int r = i >> 4, c0 = (i & 15) << 3;
    *(float4*)&sW[r][c0] = wv[i];
  }
  // stage A tile (ROWS x 128), f32 -> f16, 8B chunks -> ds_store_b64
  for (int i = tid; i < ROWS * 32; i += 256) {
    int r = i >> 5, c0 = (i & 31) << 2;
    int gr = row0 + r;
    float4 q = make_float4(0.f, 0.f, 0.f, 0.f);
    if (gr < N) q = ((const float4*)(A + (long)gr * DH))[i & 31];
    union { _Float16 h[4]; float2 f2; } u;
    u.h[0] = (_Float16)q.x; u.h[1] = (_Float16)q.y;
    u.h[2] = (_Float16)q.z; u.h[3] = (_Float16)q.w;
    *(float2*)&sA[r][c0] = u.f2;
  }
  __syncthreads();

  const int lane  = tid & 31;
  const int wcol0 = (tid >> 5) << 4;   // wave's 16-col slice
  const int rc    = lane & 15;         // A row / B col within tile
  const int hi    = lane >> 4;         // upper half-lane flag

  v8f acc0 = {}, acc1 = {}, acc2 = {}, acc3 = {};
  #pragma unroll
  for (int kk = 0; kk < 4; ++kk) {
    // B fragment: K = e + 16*hi, contiguous 16 halves (two aligned 16B loads)
    const int kbB = kk * 32 + (hi << 4);
    v8h b0 = *(const v8h*)&sW[wcol0 + rc][kbB];
    v8h b1 = *(const v8h*)&sW[wcol0 + rc][kbB + 8];
    v16h b = __builtin_shufflevector(b0, b1, 0,1,2,3,4,5,6,7,8,9,10,11,12,13,14,15);

    // A fragments: K = e + 16*hi, contiguous 16 halves (two aligned 16B loads)
    const int kbA = kk * 32 + (hi << 4);
    #pragma unroll
    for (int rt = 0; rt < 4; ++rt) {
      const _Float16* ar = &sA[rt * 16 + rc][0];
      v8h a0 = *(const v8h*)(ar + kbA);
      v8h a1 = *(const v8h*)(ar + kbA + 8);
      v16h a = __builtin_shufflevector(a0, a1, 0,1,2,3,4,5,6,7,8,9,10,11,12,13,14,15);
      v8f& acc = (rt == 0) ? acc0 : (rt == 1) ? acc1 : (rt == 2) ? acc2 : acc3;
      acc = __builtin_amdgcn_wmma_f32_16x16x32_f16(
          false, a, false, b, (short)0, acc, false, false);
    }
  }

  // epilogue: D VGPR v -> row = v + 8*hi (within tile), col = lane%16
  const int col = wcol0 + rc;
  const float bv = bias ? bias[col] : 0.0f;
  #pragma unroll
  for (int rt = 0; rt < 4; ++rt) {
    const v8f& acc = (rt == 0) ? acc0 : (rt == 1) ? acc1 : (rt == 2) ? acc2 : acc3;
    const int rb = row0 + rt * 16 + (hi << 3);
    #pragma unroll
    for (int v = 0; v < 8; ++v) {
      int gr = rb + v;
      if (gr < N) {
        float val = acc[v] + bv;
        if (act == 1) val = eluf(val);
        out[(long)gr * DH + col] = val;
      }
    }
  }
}

// ---------------- edge scatter: agg[dst] += norm * xw[src] ----------------
// one wave per 4 edges; lane handles 4 consecutive floats; prefetch next gather row
__global__ void k_agg(const float* __restrict__ xw, const int* __restrict__ src,
                      const int* __restrict__ dst, const float* __restrict__ dinv,
                      float* __restrict__ agg, int E) {
  const long wave = ((long)blockIdx.x * blockDim.x + threadIdx.x) >> 5;
  const int  l    = threadIdx.x & 31;
  long e0 = wave * 4;
  #pragma unroll
  for (int j = 0; j < 4; ++j) {
    long e = e0 + j;
    if (e >= E) return;
    int s = src[e], d = dst[e];
    if (j < 3 && e + 1 < E)  // prefetch next edge's source row (global_prefetch_b8)
      __builtin_prefetch(xw + (long)src[e + 1] * DH + l * 4, 0, 3);
    float nrm = dinv[s] * dinv[d];
    float4 v = ((const float4*)(xw + (long)s * DH))[l];
    float* o = agg + (long)d * DH + l * 4;
    unsafeAtomicAdd(o + 0, v.x * nrm);
    unsafeAtomicAdd(o + 1, v.y * nrm);
    unsafeAtomicAdd(o + 2, v.z * nrm);
    unsafeAtomicAdd(o + 3, v.w * nrm);
  }
}

// ---------------- h = act(agg + xw * dinv^2 + bias) ----------------
__global__ void k_fin(const float* __restrict__ agg, const float* __restrict__ xw,
                      const float* __restrict__ dinv, const float* __restrict__ bias,
                      float* __restrict__ out, long n, int relu) {
  long i = (long)blockIdx.x * blockDim.x + threadIdx.x;
  if (i >= n) return;
  int node = (int)(i >> 7), c = (int)(i & 127);
  float dv = dinv[node];
  float v = agg[i] + xw[i] * dv * dv + bias[c];
  if (relu) v = fmaxf(v, 0.0f);
  out[i] = v;
}

// ---------------- head: sigmoid(elu(h @ w3 + b3)), one wave per node ----------------
__global__ void k_head(const float* __restrict__ h, const float* __restrict__ w,
                       const float* __restrict__ b, float* __restrict__ out, int N) {
  long t = (long)blockIdx.x * blockDim.x + threadIdx.x;
  int node = (int)(t >> 5);
  if (node >= N) return;
  int l = (int)(t & 31);
  float4 hv = ((const float4*)(h + (long)node * DH))[l];
  float4 wv = ((const float4*)w)[l];
  float p = hv.x * wv.x + hv.y * wv.y + hv.z * wv.z + hv.w * wv.w;
  #pragma unroll
  for (int off = 16; off > 0; off >>= 1) p += __shfl_xor(p, off, 32);
  if (l == 0) {
    float v = eluf(p + b[0]);
    out[node] = 1.0f / (1.0f + __expf(-v));
  }
}

extern "C" void kernel_launch(void* const* d_in, const int* in_sizes, int n_in,
                              void* d_out, int out_size, void* d_ws, size_t ws_size,
                              hipStream_t stream) {
  const float* x = (const float*)d_in[0];
  const int*   ei = (const int*)d_in[1];
  const float* gw[3] = {(const float*)d_in[2], (const float*)d_in[4], (const float*)d_in[6]};
  const float* gb[3] = {(const float*)d_in[3], (const float*)d_in[5], (const float*)d_in[7]};
  const float* lw[4] = {(const float*)d_in[8], (const float*)d_in[10], (const float*)d_in[12], (const float*)d_in[14]};
  const float* lb[4] = {(const float*)d_in[9], (const float*)d_in[11], (const float*)d_in[13], (const float*)d_in[15]};

  const int N = in_sizes[0] / DH;
  const int E = in_sizes[1] / 2;
  const int* src = ei;
  const int* dst = ei + E;

  const long NH = (long)N * DH;
  float*    bufA = (float*)d_ws;       // ping: h
  float*    bufB = bufA + NH;          // xw / ping-pong
  float*    bufC = bufB + NH;          // agg / ping-pong
  float*    dinv = bufC + NH;          // [N] deg -> rsqrt
  _Float16* wt   = (_Float16*)(dinv + N);   // 128x128 f16 transposed weights

  dim3 blk(256);
  auto cdiv = [](long a, long b) { return (int)((a + b - 1) / b); };

  // degree normalization
  k_zero<<<cdiv(N, 256), blk, 0, stream>>>(dinv, N);
  k_deg<<<cdiv(E, 256), blk, 0, stream>>>(dst, dinv, E);
  k_dinv<<<cdiv(N, 256), blk, 0, stream>>>(dinv, N);

  const int gemmGrid = cdiv(N, ROWS);
  const int aggGrid  = cdiv(cdiv(E, 4) * 32L, 256);

  // 3x GCNConv (ReLU after first two)
  const float* in = x;
  for (int l = 0; l < 3; ++l) {
    k_w2t<<<cdiv(DH * DH, 256), blk, 0, stream>>>(gw[l], wt, DH * DH);
    k_gemm<<<gemmGrid, blk, 0, stream>>>(in, wt, nullptr, bufB, N, 0);
    k_zero<<<cdiv(NH, 256), blk, 0, stream>>>(bufC, NH);
    k_agg<<<aggGrid, blk, 0, stream>>>(bufB, src, dst, dinv, bufC, E);
    k_fin<<<cdiv(NH, 256), blk, 0, stream>>>(bufC, bufB, dinv, gb[l], bufA, NH, (l < 2) ? 1 : 0);
    in = bufA;
  }

  // 3x Linear(128,128) + ELU (fused in GEMM epilogue)
  k_w2t<<<cdiv(DH * DH, 256), blk, 0, stream>>>(lw[0], wt, DH * DH);
  k_gemm<<<gemmGrid, blk, 0, stream>>>(bufA, wt, lb[0], bufB, N, 1);
  k_w2t<<<cdiv(DH * DH, 256), blk, 0, stream>>>(lw[1], wt, DH * DH);
  k_gemm<<<gemmGrid, blk, 0, stream>>>(bufB, wt, lb[1], bufC, N, 1);
  k_w2t<<<cdiv(DH * DH, 256), blk, 0, stream>>>(lw[2], wt, DH * DH);
  k_gemm<<<gemmGrid, blk, 0, stream>>>(bufC, wt, lb[2], bufA, N, 1);

  // Linear(128,1) + ELU + sigmoid
  k_head<<<cdiv((long)N * 32, 256), blk, 0, stream>>>(bufA, lw[3], lb[3], (float*)d_out, N);
}
